// CMSIPlus_25933012533386
// MI455X (gfx1250) — matrix operand order repaired and checked by reading, not compiled, not run on hardware
//
#include <hip/hip_runtime.h>
#include <hip/hip_bf16.h>

#define NROW 8192
#define DDIM 128
#define NCHUNK 8
#define COLS_PER_CHUNK (NROW / NCHUNK)        // 1024
#define TILES_PER_CHUNK (COLS_PER_CHUNK / 16) // 64
#define MU (1.0f / 8192.0f)
// -(1/eps) * log2(e) : fold exp(-cost/eps) into a single v_exp_f32 (2^x)
#define NEG_INV_EPS_LOG2E (-28.853900817779268f)

typedef __attribute__((ext_vector_type(16))) __bf16 v16bf;
typedef __attribute__((ext_vector_type(8)))  __bf16 v8bf;
typedef __attribute__((ext_vector_type(8)))  float  v8f;

__device__ __forceinline__ v16bf load_afrag(const __bf16* base) {
    v16bf a = *(const v16bf*)(base);
    return a;
}

// ---------------------------------------------------------------------------
// Prep: fp32 -> bf16 copy + fp32 squared row norms. One block (128 thr) per row.
// ---------------------------------------------------------------------------
__global__ __launch_bounds__(128) void ot_prep(const float* __restrict__ H,
                                               __hip_bfloat16* __restrict__ Hb,
                                               float* __restrict__ sq) {
    const int r = blockIdx.x;
    const int t = threadIdx.x; // 0..127
    const float f = H[r * DDIM + t];
    Hb[r * DDIM + t] = __float2bfloat16(f);
    float s = f * f;
    s += __shfl_xor(s, 1);
    s += __shfl_xor(s, 2);
    s += __shfl_xor(s, 4);
    s += __shfl_xor(s, 8);
    s += __shfl_xor(s, 16);
    __shared__ float red[4];
    if ((t & 31) == 0) red[t >> 5] = s;
    __syncthreads();
    if (t == 0) sq[r] = red[0] + red[1] + red[2] + red[3];
}

__global__ void ot_fill(float* __restrict__ p, float v, int n) {
    int i = blockIdx.x * blockDim.x + threadIdx.x;
    if (i < n) p[i] = v;
}

// ---------------------------------------------------------------------------
// Fused matvec: y = M * x, M[r,c] = exp(-max(sqr[r]+sqc[c]-2*R_r.C_c,0)/eps).
// Never materializes M. Each wave: 32 rows (two 16-row WMMA groups sharing
// every B fragment) x one 1024-column chunk, with software-pipelined
// (double-buffered) B-fragment loads. grid.x = (NROW/256)*NCHUNK, block 256.
// Deterministic: fixed-order partials, no float atomics.
// ---------------------------------------------------------------------------
__global__ __launch_bounds__(256) void ot_matvec(const __bf16* __restrict__ R,
                                                 const float* __restrict__ sqr,
                                                 const __bf16* __restrict__ C,
                                                 const float* __restrict__ sqc,
                                                 const float* __restrict__ x,
                                                 float* __restrict__ partial) {
    const int lane    = threadIdx.x & 31;
    const int wave    = threadIdx.x >> 5;
    const int chunk   = blockIdx.x & (NCHUNK - 1);
    const int rb      = blockIdx.x >> 3;
    const int rbase   = rb * 256 + wave * 32;          // 32 rows per wave
    const int rowA    = rbase + (lane & 15);
    const int halfoff = (lane < 16) ? 0 : 8;
    const int akb     = (lane < 16) ? 0 : 16;  // A k-base within 32-k step
    const int bkb     = (lane < 16) ? 0 : 16;  // B k-base within 32-k step

    // Hoist A fragments for both row groups, all 4 k-steps.
    v16bf afA[4], afB[4];
#pragma unroll
    for (int ks = 0; ks < 4; ++ks) {
        afA[ks] = load_afrag(R + rowA * DDIM + ks * 32 + akb);
        afB[ks] = load_afrag(R + (rowA + 16) * DDIM + ks * 32 + akb);
    }
    float sqA[8], sqB[8], rA[8], rB[8];
#pragma unroll
    for (int e = 0; e < 8; ++e) {
        sqA[e] = sqr[rbase + e + halfoff];
        sqB[e] = sqr[rbase + 16 + e + halfoff];
        rA[e] = 0.0f;
        rB[e] = 0.0f;
    }

    const int jstart = chunk * COLS_PER_CHUNK;
    const int lcol   = lane & 15;

    // Preload B fragments + per-column scalars for tile 0.
    v16bf bcur[4];
    {
        const int col = jstart + lcol;
#pragma unroll
        for (int ks = 0; ks < 4; ++ks)
            bcur[ks] = *(const v16bf*)(C + col * DDIM + ks * 32 + bkb);
    }
    float sc = sqc[jstart + lcol];
    float xj = x[jstart + lcol];

    for (int t = 0; t < TILES_PER_CHUNK; ++t) {
        // Issue next tile's loads first (wraps at the end; redundant but
        // branch-free and cache-resident) so they overlap WMMA + exp work.
        const int coln = jstart + (((t + 1) & (TILES_PER_CHUNK - 1)) << 4) + lcol;
        v16bf bnxt[4];
#pragma unroll
        for (int ks = 0; ks < 4; ++ks)
            bnxt[ks] = *(const v16bf*)(C + coln * DDIM + ks * 32 + bkb);
        const float scn = sqc[coln];
        const float xn  = x[coln];

        v8f accA = {}, accB = {};
#pragma unroll
        for (int ks = 0; ks < 4; ++ks) {
            accA = __builtin_amdgcn_wmma_f32_16x16x32_bf16(
                false, afA[ks], false, bcur[ks], (short)0, accA, false, false);
            accB = __builtin_amdgcn_wmma_f32_16x16x32_bf16(
                false, afB[ks], false, bcur[ks], (short)0, accB, false, false);
        }
#pragma unroll
        for (int e = 0; e < 8; ++e) {
            float dA = fmaf(-2.0f, accA[e], sqA[e] + sc);
            float dB = fmaf(-2.0f, accB[e], sqB[e] + sc);
            dA = fmaxf(dA, 0.0f);
            dB = fmaxf(dB, 0.0f);
            rA[e] = fmaf(__builtin_amdgcn_exp2f(dA * NEG_INV_EPS_LOG2E), xj, rA[e]);
            rB[e] = fmaf(__builtin_amdgcn_exp2f(dB * NEG_INV_EPS_LOG2E), xj, rB[e]);
        }
#pragma unroll
        for (int ks = 0; ks < 4; ++ks) bcur[ks] = bnxt[ks];
        sc = scn;
        xj = xn;
    }

    // Row sums: reduce across each 16-lane group (columns live across lanes).
#pragma unroll
    for (int e = 0; e < 8; ++e) {
        float sA = rA[e], sB = rB[e];
        sA += __shfl_xor(sA, 1); sB += __shfl_xor(sB, 1);
        sA += __shfl_xor(sA, 2); sB += __shfl_xor(sB, 2);
        sA += __shfl_xor(sA, 4); sB += __shfl_xor(sB, 4);
        sA += __shfl_xor(sA, 8); sB += __shfl_xor(sB, 8);
        if ((lane & 15) == 0) {
            partial[chunk * NROW + rbase + e + halfoff]      = sA;
            partial[chunk * NROW + rbase + 16 + e + halfoff] = sB;
        }
    }
}

// out[i] = numer / sum_chunks(partial[c][i])  — the Sinkhorn divide.
__global__ void ot_reduce(const float* __restrict__ partial,
                          float* __restrict__ out, float numer) {
    int i = blockIdx.x * blockDim.x + threadIdx.x;
    if (i < NROW) {
        float s = 0.0f;
#pragma unroll
        for (int c = 0; c < NCHUNK; ++c) s += partial[c * NROW + i];
        out[i] = numer / s;
    }
}

// ---------------------------------------------------------------------------
// Final plan: out[i,j] = u[i]*v[j]*exp(-cost/eps), recomputed via WMMA tiles.
// Each wave: 16 rows x 128 cols, double-buffered B, nontemporal 256MB store.
// grid.x = 512 * 8, block = 256.
// ---------------------------------------------------------------------------
__global__ __launch_bounds__(256) void ot_plan(const __bf16* __restrict__ Am,
                                               const float* __restrict__ sqm,
                                               const __bf16* __restrict__ An,
                                               const float* __restrict__ sqn,
                                               const float* __restrict__ u,
                                               const float* __restrict__ v,
                                               float* __restrict__ out) {
    const int lane    = threadIdx.x & 31;
    const int wave    = threadIdx.x >> 5;
    const int rb      = blockIdx.x & 511;  // 512 row tiles
    const int cb      = blockIdx.x >> 9;   // 8 column super-blocks
    const int rbase   = rb * 16;
    const int row     = rbase + (lane & 15);
    const int halfoff = (lane < 16) ? 0 : 8;
    const int akb     = (lane < 16) ? 0 : 16;
    const int bkb     = (lane < 16) ? 0 : 16;

    v16bf afrag[4];
#pragma unroll
    for (int ks = 0; ks < 4; ++ks)
        afrag[ks] = load_afrag(Am + row * DDIM + ks * 32 + akb);

    float sqr_r[8], u_r[8];
#pragma unroll
    for (int e = 0; e < 8; ++e) {
        const int idx = rbase + e + halfoff;
        sqr_r[e] = sqm[idx];
        u_r[e]   = u[idx];
    }

    const int cstart = cb * 1024 + wave * 128;
    const int lcol   = lane & 15;

    v16bf bcur[4];
#pragma unroll
    for (int ks = 0; ks < 4; ++ks)
        bcur[ks] = *(const v16bf*)(An + (cstart + lcol) * DDIM + ks * 32 + bkb);
    float sc = sqn[cstart + lcol];
    float vj = v[cstart + lcol];

    for (int t = 0; t < 8; ++t) {
        const int col  = cstart + t * 16 + lcol;
        const int coln = cstart + (((t + 1) & 7) << 4) + lcol;
        v16bf bnxt[4];
#pragma unroll
        for (int ks = 0; ks < 4; ++ks)
            bnxt[ks] = *(const v16bf*)(An + coln * DDIM + ks * 32 + bkb);
        const float scn = sqn[coln];
        const float vn  = v[coln];

        v8f acc = {};
#pragma unroll
        for (int ks = 0; ks < 4; ++ks)
            acc = __builtin_amdgcn_wmma_f32_16x16x32_bf16(
                false, afrag[ks], false, bcur[ks], (short)0, acc, false, false);

#pragma unroll
        for (int e = 0; e < 8; ++e) {
            float d = fmaf(-2.0f, acc[e], sqr_r[e] + sc);
            d = fmaxf(d, 0.0f);
            const float val =
                u_r[e] * vj * __builtin_amdgcn_exp2f(d * NEG_INV_EPS_LOG2E);
            __builtin_nontemporal_store(
                val, &out[(size_t)(rbase + e + halfoff) * NROW + col]);
        }
#pragma unroll
        for (int ks = 0; ks < 4; ++ks) bcur[ks] = bnxt[ks];
        sc = scn;
        vj = vn;
    }
}

// ---------------------------------------------------------------------------
extern "C" void kernel_launch(void* const* d_in, const int* in_sizes, int n_in,
                              void* d_out, int out_size, void* d_ws, size_t ws_size,
                              hipStream_t stream) {
    const float* Hm = (const float*)d_in[0];
    const float* Hn = (const float*)d_in[1];
    float* out = (float*)d_out;

    char* ws = (char*)d_ws;
    const size_t BF_BYTES = (size_t)NROW * DDIM * 2; // 2 MB each
    __hip_bfloat16* AmW = (__hip_bfloat16*)(ws);
    __hip_bfloat16* AnW = (__hip_bfloat16*)(ws + BF_BYTES);
    const __bf16* Am = (const __bf16*)AmW;
    const __bf16* An = (const __bf16*)AnW;
    float* sqm     = (float*)(ws + 2 * BF_BYTES);
    float* sqn     = sqm + NROW;
    float* u       = sqn + NROW;
    float* w       = u + NROW;
    float* partial = w + NROW; // NCHUNK * NROW floats

    // Precision prep: bf16 operands + fp32 norms.
    ot_prep<<<NROW, 128, 0, stream>>>(Hm, AmW, sqm);
    ot_prep<<<NROW, 128, 0, stream>>>(Hn, AnW, sqn);
    ot_fill<<<NROW / 256, 256, 0, stream>>>(u, 1.0f, NROW);

    const int mvGrid = (NROW / 256) * NCHUNK; // 256 blocks
    for (int it = 0; it < 20; ++it) {
        // t = K^T u  (rows = Hn side, cols = Hm side);  w = nu / t
        ot_matvec<<<mvGrid, 256, 0, stream>>>(An, sqn, Am, sqm, u, partial);
        ot_reduce<<<NROW / 256, 256, 0, stream>>>(partial, w, MU);
        // a = K w;  u = mu / a
        ot_matvec<<<mvGrid, 256, 0, stream>>>(Am, sqm, An, sqn, w, partial);
        ot_reduce<<<NROW / 256, 256, 0, stream>>>(partial, u, MU);
    }
    // v = nu / (K^T u)
    ot_matvec<<<mvGrid, 256, 0, stream>>>(An, sqn, Am, sqm, u, partial);
    ot_reduce<<<NROW / 256, 256, 0, stream>>>(partial, w, MU);

    // plan = u v^T ∘ K
    ot_plan<<<512 * 8, 256, 0, stream>>>(Am, sqm, An, sqn, u, w, out);
}
